// Model_81990925681018
// MI455X (gfx1250) — matrix operand order repaired and checked by reading, not compiled, not run on hardware
//
#include <hip/hip_runtime.h>
#include <stdint.h>

// ---------------------------------------------------------------------------
// CDNA5 (gfx1250) implementation: all matmuls via v_wmma_f32_16x16x32_bf16.
// Activations token-major [rows][K_pad] bf16 so WMMA fragment loads are
// contiguous 16B/32B chunks per lane. Conv kernels templated on (Cin_pad, k)
// so im2col tap math is shifts, with a scalar interior/edge split so the hot
// path has no EXEC predication. K loops unrolled x2 with dual accumulators.
// ---------------------------------------------------------------------------

typedef unsigned short u16;
typedef __attribute__((ext_vector_type(16))) __bf16 v16bf;
typedef __attribute__((ext_vector_type(8)))  float  v8f;

union Frag { v16bf v; uint4 q[2]; };

__device__ __forceinline__ u16 f2bf(float f) {
    union { float f; unsigned int u; } v; v.f = f;
    unsigned int u = v.u;
    u += 0x7fffu + ((u >> 16) & 1u);   // round-to-nearest-even
    return (u16)(u >> 16);
}
__device__ __forceinline__ float bf2f(u16 h) {
    union { unsigned int u; float f; } v; v.u = ((unsigned int)h) << 16;
    return v.f;
}
__device__ __forceinline__ v8f wmma_bf16(v16bf a, v16bf b, v8f c) {
    return __builtin_amdgcn_wmma_f32_16x16x32_bf16(false, a, false, b, (short)0, c,
                                                   false, false);
}
__device__ __forceinline__ v8f vzero8() {
    v8f z = {0.f, 0.f, 0.f, 0.f, 0.f, 0.f, 0.f, 0.f};
    return z;
}

// --------------------------- small utility kernels -------------------------

__global__ void k_gather_rows(const int* __restrict__ idx,
                              const float* __restrict__ emb,
                              u16* __restrict__ out) {
    int r = blockIdx.x, t = threadIdx.x;          // 256 threads, D=256
    int e = idx[r];
    out[(size_t)r * 256 + t] = f2bf(emb[(size_t)e * 256 + t]);
}

__global__ void k_pack_conv(const float* __restrict__ w, u16* __restrict__ out,
                            int F_, int Cin, int ksz, int Cinp, int Mpad) {
    long Ktot = (long)Cinp * ksz;
    long total = (long)Mpad * Ktot;
    long i = (long)blockIdx.x * blockDim.x + threadIdx.x;
    if (i >= total) return;
    int m = (int)(i / Ktot), kk = (int)(i % Ktot);
    int tap = kk / Cinp, c = kk % Cinp;
    float v = 0.f;
    if (m < F_ && c < Cin) v = w[((size_t)m * Cin + c) * ksz + tap];
    out[i] = f2bf(v);
}

__global__ void k_pack_mat(const float* __restrict__ w, u16* __restrict__ out,
                           int M_, int K_, int Mp, int Kp) {
    long total = (long)Mp * Kp;
    long i = (long)blockIdx.x * blockDim.x + threadIdx.x;
    if (i >= total) return;
    int m = (int)(i / Kp), k = (int)(i % Kp);
    float v = (m < M_ && k < K_) ? w[(size_t)m * K_ + k] : 0.f;
    out[i] = f2bf(v);
}

__global__ void k_fill_f32(float* p, float v, long n) {
    long i = (long)blockIdx.x * blockDim.x + threadIdx.x;
    if (i < n) p[i] = v;
}
__global__ void k_zero_u16(u16* p, long n) {
    long i = (long)blockIdx.x * blockDim.x + threadIdx.x;
    if (i < n) p[i] = 0;
}

// ------------------------- WMMA conv (im2col) GEMM -------------------------
// in  : [B][L][CINP] bf16   W : [128][CINP*KSZ] bf16 packed
// out : [B][L][128]  bf16.  MODE 0: tanh(conv+bias) then BN (conv1)
//                           MODE 1: tanh(BN(conv) + resid)   (residual)
template <int CINP, int KSZ, int MODE>
__global__ __launch_bounds__(32)
void k_conv_gemm(const u16* __restrict__ in, const u16* __restrict__ W,
                 u16* __restrict__ out, const u16* __restrict__ resid,
                 const float* __restrict__ bias,
                 const float* __restrict__ gamma, const float* __restrict__ beta,
                 int L_, int Fdim) {
    constexpr int KTOT = CINP * KSZ;          // multiple of 64
    constexpr int KHALF = KSZ / 2;
    const int lane = threadIdx.x & 31;
    const int g = lane >> 4, ln = lane & 15;
    const int Lbase = blockIdx.x * 16;
    const int Mbase = blockIdx.y * 16;
    const int b = blockIdx.z;
    const u16* wrow = W + (size_t)(Mbase + ln) * KTOT;
    const u16* irow = in + (size_t)b * L_ * CINP;
    v8f acc;

    // interior tiles never touch the 'same' padding: uniform scalar branch
    if (blockIdx.x != 0 && blockIdx.x != gridDim.x - 1) {
        v8f acc0 = vzero8(), acc1 = vzero8();
        for (int kk = 0; kk < KTOT; kk += 64) {
            Frag fa0, fb0, fa1, fb1;
            fa0.q[0] = *(const uint4*)(wrow + kk + g * 16);
            fa0.q[1] = *(const uint4*)(wrow + kk + 8 + g * 16);
            fa1.q[0] = *(const uint4*)(wrow + kk + 32 + g * 16);
            fa1.q[1] = *(const uint4*)(wrow + kk + 40 + g * 16);
            {
                int kel = kk + g * 16;                      // B: K = kel..kel+15
                int tap = kel >> __builtin_ctz(CINP);       // constant shift
                int coff = kel & (CINP - 1);
                const uint4* bp =
                    (const uint4*)(irow + (size_t)(Lbase + ln + tap - KHALF) * CINP + coff);
                fb0.q[0] = bp[0]; fb0.q[1] = bp[1];
            }
            {
                int kel = kk + 32 + g * 16;
                int tap = kel >> __builtin_ctz(CINP);
                int coff = kel & (CINP - 1);
                const uint4* bp =
                    (const uint4*)(irow + (size_t)(Lbase + ln + tap - KHALF) * CINP + coff);
                fb1.q[0] = bp[0]; fb1.q[1] = bp[1];
            }
            acc0 = wmma_bf16(fa0.v, fb0.v, acc0);
            acc1 = wmma_bf16(fa1.v, fb1.v, acc1);
        }
        acc = acc0 + acc1;
    } else {
        acc = vzero8();
        for (int kk = 0; kk < KTOT; kk += 32) {
            Frag fa, fb;
            fa.q[0] = *(const uint4*)(wrow + kk + g * 16);
            fa.q[1] = *(const uint4*)(wrow + kk + 8 + g * 16);
            int kel = kk + g * 16;
            int tap = kel >> __builtin_ctz(CINP);
            int coff = kel & (CINP - 1);
            int l = Lbase + ln + tap - KHALF;
            if ((unsigned)l < (unsigned)L_) {
                const uint4* bp = (const uint4*)(irow + (size_t)l * CINP + coff);
                fb.q[0] = bp[0]; fb.q[1] = bp[1];
            } else {
                fb.q[0] = make_uint4(0, 0, 0, 0);
                fb.q[1] = make_uint4(0, 0, 0, 0);
            }
            acc = wmma_bf16(fa.v, fb.v, acc);
        }
    }

    const int l = Lbase + ln;
    const size_t obase = ((size_t)b * L_ + l) * 128;
    const float bnrs = 0.999995000037497f;   // 1/sqrt(1+1e-5)
#pragma unroll
    for (int r = 0; r < 8; ++r) {
        int m = Mbase + 8 * g + r;           // C layout: VGPR r -> M = 8g+r, N = ln
        float v = 0.f;
        if (m < Fdim) {
            v = acc[r];
            float sc = gamma[m] * bnrs;
            if (MODE == 0) {
                v = tanhf(v + bias[m]);
                v = v * sc + beta[m];
            } else {
                v = v * sc + beta[m] + bf2f(resid[obase + m]);
                v = tanhf(v);
            }
        }
        out[obase + m] = f2bf(v);
    }
}

// --------------------------- generic WMMA GEMM -----------------------------
// C[m][n] = sum_k A[m][k]*Bt[n][k];  A:[Mpad][K] bf16, Bt:[Nrows][K] bf16.
// out index = bt*oBS + m*oMS + n*oNS ; act: 0 none, 1 tanh, 2 relu.
// K_ must be a multiple of 64.
__global__ __launch_bounds__(32)
void k_gemm(const u16* __restrict__ A, const u16* __restrict__ Bt, int K_,
            long long aBS, long long bBS,
            float* __restrict__ outF, u16* __restrict__ outH,
            long long oBS, long long oMS, long long oNS,
            const float* __restrict__ bias, int act) {
    const int lane = threadIdx.x & 31;
    const int g = lane >> 4, ln = lane & 15;
    const int Nbase = blockIdx.x * 16;
    const int Mbase = blockIdx.y * 16;
    const int bt = blockIdx.z;
    const u16* arow = A + (size_t)bt * aBS + (size_t)(Mbase + ln) * K_;
    const u16* brow = Bt + (size_t)bt * bBS + (size_t)(Nbase + ln) * K_;
    v8f acc0 = vzero8(), acc1 = vzero8();
    for (int kk = 0; kk < K_; kk += 64) {
        Frag fa0, fb0, fa1, fb1;
        fa0.q[0] = *(const uint4*)(arow + kk + g * 16);
        fa0.q[1] = *(const uint4*)(arow + kk + 8 + g * 16);
        fa1.q[0] = *(const uint4*)(arow + kk + 32 + g * 16);
        fa1.q[1] = *(const uint4*)(arow + kk + 40 + g * 16);
        const uint4* bp0 = (const uint4*)(brow + kk + g * 16);
        const uint4* bp1 = (const uint4*)(brow + kk + 32 + g * 16);
        fb0.q[0] = bp0[0]; fb0.q[1] = bp0[1];
        fb1.q[0] = bp1[0]; fb1.q[1] = bp1[1];
        acc0 = wmma_bf16(fa0.v, fb0.v, acc0);
        acc1 = wmma_bf16(fa1.v, fb1.v, acc1);
    }
    v8f acc = acc0 + acc1;
    const int n = Nbase + ln;
#pragma unroll
    for (int r = 0; r < 8; ++r) {
        int m = Mbase + 8 * g + r;
        float v = acc[r];
        if (bias) v += bias[m];
        if (act == 1) v = tanhf(v);
        else if (act == 2) v = fmaxf(v, 0.f);
        size_t idx = (size_t)bt * oBS + (size_t)m * oMS + (size_t)n * oNS;
        if (outF) outF[idx] = v;
        else      outH[idx] = f2bf(v);
    }
}

// ------------------------------- misc stages -------------------------------

__global__ void k_transpose_LF(const u16* __restrict__ in, u16* __restrict__ out,
                               int L_, int Fp) {
    int bl = blockIdx.x, f = threadIdx.x;      // block Fp threads
    int b = bl / L_, l = bl % L_;
    out[((size_t)b * Fp + f) * L_ + l] = in[(size_t)bl * Fp + f];
}

__global__ void k_softmax(const float* __restrict__ S, u16* __restrict__ Wt,
                          int L_, int Cp) {
    __shared__ float red[256];
    int bc = blockIdx.x, b = bc / Cp, c = bc % Cp;
    int t = threadIdx.x;
    const float* sp = S + (size_t)b * L_ * Cp + c;
    float mx = -3.4e38f;
    for (int l = t; l < L_; l += 256) mx = fmaxf(mx, sp[(size_t)l * Cp]);
    red[t] = mx; __syncthreads();
    for (int s = 128; s > 0; s >>= 1) { if (t < s) red[t] = fmaxf(red[t], red[t + s]); __syncthreads(); }
    mx = red[0]; __syncthreads();
    float sm = 0.f;
    for (int l = t; l < L_; l += 256) sm += expf(sp[(size_t)l * Cp] - mx);
    red[t] = sm; __syncthreads();
    for (int s = 128; s > 0; s >>= 1) { if (t < s) red[t] += red[t + s]; __syncthreads(); }
    float inv = 1.f / red[0];
    u16* wp = Wt + ((size_t)b * Cp + c) * L_;
    for (int l = t; l < L_; l += 256) wp[l] = f2bf(expf(sp[(size_t)l * Cp] - mx) * inv);
}

__global__ void k_deg(const int* __restrict__ edges, float* __restrict__ deg,
                      int E_, int Nn_, long total) {
    long i = (long)blockIdx.x * blockDim.x + threadIdx.x;
    if (i >= total) return;
    int b = (int)(i / E_), e = (int)(i % E_);
    int d = edges[((size_t)b * 2 + 1) * E_ + e];
    atomicAdd(&deg[(size_t)b * Nn_ + d], 1.0f);
}

__global__ void k_scatter(const int* __restrict__ edges, const float* __restrict__ deg,
                          const float* __restrict__ xw, float* __restrict__ agg,
                          int E_, int Nn_) {
    int b = blockIdx.x / (E_ + Nn_);
    int e = blockIdx.x % (E_ + Nn_);
    int s, d;
    if (e < E_) {
        s = edges[((size_t)b * 2) * E_ + e];
        d = edges[((size_t)b * 2 + 1) * E_ + e];
    } else { s = d = e - E_; }                 // self loops
    float nm = rsqrtf(deg[(size_t)b * Nn_ + s]) * rsqrtf(deg[(size_t)b * Nn_ + d]);
    int t = threadIdx.x;                       // 256 channels
    atomicAdd(&agg[((size_t)b * Nn_ + d) * 256 + t],
              xw[((size_t)b * Nn_ + s) * 256 + t] * nm);
}

__global__ void k_post1(const float* __restrict__ agg, const float* __restrict__ bias,
                        u16* __restrict__ out) {
    long r = blockIdx.x; int t = threadIdx.x;
    float v = fmaxf(agg[r * 256 + t] + bias[t], 0.f);
    out[r * 256 + t] = f2bf(v);
}

__global__ void k_post2(const float* __restrict__ agg, const float* __restrict__ bias,
                        u16* __restrict__ conc, int Nn_, int Cc_) {
    int b = blockIdx.x / Cc_, c = blockIdx.x % Cc_; int t = threadIdx.x;
    float v = agg[((size_t)b * Nn_ + c) * 256 + t] + bias[t];
    conc[((size_t)b * Cc_ + c) * 576 + t] = f2bf(v);
}

__global__ void k_conc_v(const float* __restrict__ V, u16* __restrict__ conc,
                         int Cc_, int Cp_, int Fp_, int B_) {
    int b = blockIdx.x / Cc_, c = blockIdx.x % Cc_;
    int j = threadIdx.x;
    if (j >= 300) return;
    int enc = j / 100, f = j % 100;
    float v = V[(((size_t)enc * B_ + b) * Cp_ + c) * Fp_ + f];
    conc[((size_t)b * Cc_ + c) * 576 + 256 + j] = f2bf(v);
}

__global__ void k_fin2(const float* __restrict__ hid, const float* __restrict__ w2,
                       const float* __restrict__ b2, float* __restrict__ y,
                       int Cc_, int FFN_) {
    __shared__ float red[128];
    int bc = blockIdx.x; int c = bc % Cc_; int t = threadIdx.x;
    red[t] = w2[(size_t)c * FFN_ + t] * hid[(size_t)bc * FFN_ + t];
    __syncthreads();
    for (int s = 64; s > 0; s >>= 1) { if (t < s) red[t] += red[t + s]; __syncthreads(); }
    if (t == 0) y[bc] = red[0] + b2[c];
}

// ----------------------- conv template dispatcher --------------------------

static void launch_conv(const u16* in, const u16* W, u16* out, const u16* resid,
                        const float* bias, const float* gamma, const float* beta,
                        int k, int mode, int L, int F, int B, hipStream_t s) {
    dim3 grd(L / 16, 8, B);
    if (mode == 0) {               // conv1: CINP = 256
        if (k == 3)      k_conv_gemm<256, 3, 0><<<grd, 32, 0, s>>>(in, W, out, resid, bias, gamma, beta, L, F);
        else if (k == 5) k_conv_gemm<256, 5, 0><<<grd, 32, 0, s>>>(in, W, out, resid, bias, gamma, beta, L, F);
        else             k_conv_gemm<256, 9, 0><<<grd, 32, 0, s>>>(in, W, out, resid, bias, gamma, beta, L, F);
    } else {                       // residual: CINP = 128
        if (k == 3)      k_conv_gemm<128, 3, 1><<<grd, 32, 0, s>>>(in, W, out, resid, bias, gamma, beta, L, F);
        else if (k == 5) k_conv_gemm<128, 5, 1><<<grd, 32, 0, s>>>(in, W, out, resid, bias, gamma, beta, L, F);
        else             k_conv_gemm<128, 9, 1><<<grd, 32, 0, s>>>(in, W, out, resid, bias, gamma, beta, L, F);
    }
}

// ------------------------------ host launcher ------------------------------

extern "C" void kernel_launch(void* const* d_in, const int* in_sizes, int n_in,
                              void* d_out, int out_size, void* d_ws, size_t ws_size,
                              hipStream_t stream) {
    (void)in_sizes; (void)n_in; (void)out_size; (void)ws_size;
    const int B = 16, L = 4096, D = 256, F = 100, Fp = 128, Aa = 256;
    const int Cc = 50, Cp = 64, Nn = 512, E = 8192, FFN = 128, CONC = 576;

    const int*   x          = (const int*)d_in[0];
    const int*   nodes      = (const int*)d_in[2];
    const int*   edges      = (const int*)d_in[3];
    const float* word_emb   = (const float*)d_in[4];
    const float* entity_emb = (const float*)d_in[5];
    const float* convw[3]   = {(const float*)d_in[6], (const float*)d_in[9],  (const float*)d_in[12]};
    const float* convb[3]   = {(const float*)d_in[7], (const float*)d_in[10], (const float*)d_in[13]};
    const float* resw[3]    = {(const float*)d_in[8], (const float*)d_in[11], (const float*)d_in[14]};
    const float* bn_gamma   = (const float*)d_in[15];
    const float* bn_beta    = (const float*)d_in[16];
    const float* attn_w     = (const float*)d_in[17];
    const float* attn_u     = (const float*)d_in[18];
    const float* gc1_w = (const float*)d_in[19]; const float* gc1_b = (const float*)d_in[20];
    const float* gc2_w = (const float*)d_in[21]; const float* gc2_b = (const float*)d_in[22];
    const float* fin1_w = (const float*)d_in[23]; const float* fin1_b = (const float*)d_in[24];
    const float* fin2_w = (const float*)d_in[25]; const float* fin2_b = (const float*)d_in[26];
    const int ks[3] = {3, 5, 9};

    char* wsb = (char*)d_ws;
    size_t off = 0;
    auto alloc = [&](size_t bytes) {
        void* p = wsb + off;
        off = (off + bytes + 255) & ~(size_t)255;
        return p;
    };
    u16*   h_bf  = (u16*)alloc((size_t)B * L * D * 2);
    u16*   t0    = (u16*)alloc((size_t)B * L * Fp * 2);
    u16*   t1    = (u16*)alloc((size_t)B * L * Fp * 2);
    u16*   tT    = (u16*)alloc((size_t)B * Fp * L * 2);
    u16*   Zb    = (u16*)alloc((size_t)B * L * Aa * 2);
    float* Sf    = (float*)alloc((size_t)B * L * Cp * 4);
    u16*   attnW = (u16*)alloc((size_t)B * Cp * L * 2);
    float* Vf    = (float*)alloc((size_t)3 * B * Cp * Fp * 4);
    u16*   wc    = (u16*)alloc((size_t)128 * D * 9 * 2);
    u16*   wr    = (u16*)alloc((size_t)128 * Fp * 9 * 2);
    u16*   awp   = (u16*)alloc((size_t)Aa * Fp * 2);
    u16*   aup   = (u16*)alloc((size_t)Cp * Aa * 2);
    u16*   fin1p = (u16*)alloc((size_t)FFN * CONC * 2);
    u16*   gc1p  = (u16*)alloc((size_t)D * D * 2);
    u16*   gc2p  = (u16*)alloc((size_t)D * D * 2);
    u16*   neb   = (u16*)alloc((size_t)B * Nn * D * 2);
    float* xw    = (float*)alloc((size_t)B * Nn * D * 4);
    float* agg   = (float*)alloc((size_t)B * Nn * D * 4);
    float* deg   = (float*)alloc((size_t)B * Nn * 4);
    u16*   g1b   = (u16*)alloc((size_t)B * Nn * D * 2);
    u16*   conc  = (u16*)alloc((size_t)B * Cc * CONC * 2);
    float* hid   = (float*)alloc((size_t)B * Cc * FFN * 4);

    // embeddings -> bf16
    k_gather_rows<<<B * L, 256, 0, stream>>>(x, word_emb, h_bf);
    k_gather_rows<<<B * Nn, 256, 0, stream>>>(nodes, entity_emb, neb);

    // ----- three CNN encoders -----
    for (int enc = 0; enc < 3; ++enc) {
        int k = ks[enc];
        long tot1 = 128L * D * k;
        k_pack_conv<<<(unsigned)((tot1 + 255) / 256), 256, 0, stream>>>(
            convw[enc], wc, F, D, k, D, 128);
        launch_conv(h_bf, wc, t0, nullptr, convb[enc],
                    bn_gamma + (size_t)(enc * 3 + 0) * F,
                    bn_beta + (size_t)(enc * 3 + 0) * F, k, 0, L, F, B, stream);
        u16* tin = t0; u16* tout = t1;
        for (int i = 0; i < 2; ++i) {
            long totr = 128L * Fp * k;
            k_pack_conv<<<(unsigned)((totr + 255) / 256), 256, 0, stream>>>(
                resw[enc] + (size_t)i * F * F * k, wr, F, F, k, Fp, 128);
            launch_conv(tin, wr, tout, tin, nullptr,
                        bn_gamma + (size_t)(enc * 3 + i + 1) * F,
                        bn_beta + (size_t)(enc * 3 + i + 1) * F, k, 1, L, F, B, stream);
            u16* tmp = tin; tin = tout; tout = tmp;
        }
        u16* tcur = tin;   // final encoder features [B][L][128]

        // Z = tanh(aw @ t) : out Zb[(b*L+l)][256] bf16
        k_pack_mat<<<(Aa * Fp + 255) / 256, 256, 0, stream>>>(
            attn_w + (size_t)enc * Aa * F, awp, Aa, F, Aa, Fp);
        k_gemm<<<dim3(B * L / 16, Aa / 16, 1), 32, 0, stream>>>(
            awp, tcur, Fp, 0, 0, nullptr, Zb, 0, 1, Aa, nullptr, 1);

        // S = au @ Z : out Sf[(b*L+l)][64] f32
        k_pack_mat<<<(Cp * Aa + 255) / 256, 256, 0, stream>>>(
            attn_u + (size_t)enc * Cc * Aa, aup, Cc, Aa, Cp, Aa);
        k_gemm<<<dim3(B * L / 16, Cp / 16, 1), 32, 0, stream>>>(
            aup, Zb, Aa, 0, 0, Sf, nullptr, 0, 1, Cp, nullptr, 0);

        // softmax over L -> attnW[b][c][L] bf16
        k_softmax<<<B * Cp, 256, 0, stream>>>(Sf, attnW, L, Cp);

        // t -> tT [B][128][L] for V GEMM B-operand
        k_transpose_LF<<<B * L, Fp, 0, stream>>>(tcur, tT, L, Fp);

        // V[b][c][f] = sum_l attnW[b][c][l] * t[b][f][l]
        k_gemm<<<dim3(Fp / 16, Cp / 16, B), 32, 0, stream>>>(
            attnW, tT, L, (long long)Cp * L, (long long)Fp * L,
            Vf + (size_t)enc * B * Cp * Fp, nullptr,
            (long long)Cp * Fp, Fp, 1, nullptr, 0);
    }

    // ----- GCN -----
    k_fill_f32<<<(B * Nn + 255) / 256, 256, 0, stream>>>(deg, 1.0f, (long)B * Nn);
    k_deg<<<(B * E + 255) / 256, 256, 0, stream>>>(edges, deg, E, Nn, (long)B * E);
    k_pack_mat<<<(D * D + 255) / 256, 256, 0, stream>>>(gc1_w, gc1p, D, D, D, D);
    k_pack_mat<<<(D * D + 255) / 256, 256, 0, stream>>>(gc2_w, gc2p, D, D, D, D);

    // xw = ne @ W1^T : [b][node][dout] f32
    k_gemm<<<dim3(Nn / 16, D / 16, B), 32, 0, stream>>>(
        gc1p, neb, D, 0, (long long)Nn * D, xw, nullptr,
        (long long)Nn * D, 1, D, nullptr, 0);
    k_fill_f32<<<(unsigned)(((long)B * Nn * D + 255) / 256), 256, 0, stream>>>(
        agg, 0.f, (long)B * Nn * D);
    k_scatter<<<B * (E + Nn), 256, 0, stream>>>(edges, deg, xw, agg, E, Nn);
    k_post1<<<B * Nn, 256, 0, stream>>>(agg, gc1_b, g1b);

    k_gemm<<<dim3(Nn / 16, D / 16, B), 32, 0, stream>>>(
        gc2p, g1b, D, 0, (long long)Nn * D, xw, nullptr,
        (long long)Nn * D, 1, D, nullptr, 0);
    k_fill_f32<<<(unsigned)(((long)B * Nn * D + 255) / 256), 256, 0, stream>>>(
        agg, 0.f, (long)B * Nn * D);
    k_scatter<<<B * (E + Nn), 256, 0, stream>>>(edges, deg, xw, agg, E, Nn);

    // ----- concat + final FFN -----
    k_zero_u16<<<(unsigned)(((long)B * Cc * CONC + 255) / 256), 256, 0, stream>>>(
        conc, (long)B * Cc * CONC);
    k_post2<<<B * Cc, 256, 0, stream>>>(agg, gc2_b, conc, Nn, Cc);
    k_conc_v<<<B * Cc, 512, 0, stream>>>(Vf, conc, Cc, Cp, Fp, B);

    k_pack_mat<<<(FFN * CONC + 255) / 256, 256, 0, stream>>>(
        fin1_w, fin1p, FFN, 3 * F + D, FFN, CONC);
    k_gemm<<<dim3(B * Cc / 16, FFN / 16, 1), 32, 0, stream>>>(
        fin1p, conc, CONC, 0, 0, hid, nullptr, 0, 1, FFN, fin1_b, 2);

    k_fin2<<<B * Cc, 128, 0, stream>>>(hid, fin2_w, fin2_b, (float*)d_out, Cc, FFN);
}
